// Net_66760971649748
// MI455X (gfx1250) — matrix operand order repaired and checked by reading, not compiled, not run on hardware
//
#include <hip/hip_runtime.h>
#include <hip/hip_bf16.h>

typedef __attribute__((ext_vector_type(16))) _Float16 v16h;
typedef __attribute__((ext_vector_type(8)))  _Float16 v8h;
typedef __attribute__((ext_vector_type(8)))  float    v8f;

#define BB    4
#define NN    4096
#define KNBR  15
#define NPTS  (BB*NN)       /* 16384  */
#define MEDGE (NPTS*KNBR)   /* 245760 */
#define BN_EPS 1e-5f

// ---------------------------------------------------------------------------
// WMMA fragment helpers (CDNA5 16x16x32 f16, wave32; lane half h holds contiguous k 16h..16h+15)
// ---------------------------------------------------------------------------
static __device__ __forceinline__ v16h cat16(v8h lo, v8h hi) {
  return __builtin_shufflevector(lo, hi, 0,1,2,3,4,5,6,7,8,9,10,11,12,13,14,15);
}
// A fragment from per-lane pointer: 16 contiguous halfs
static __device__ __forceinline__ v16h lda(const _Float16* p) {
  return cat16(*(const v8h*)p, *(const v8h*)(p + 8));
}
// B fragment from per-lane pointer: 16 contiguous halfs
static __device__ __forceinline__ v16h ldb(const _Float16* p) {
  return cat16(*(const v8h*)p, *(const v8h*)(p + 8));
}
static __device__ __forceinline__ v8f wmma(v16h a, v16h b, v8f c) {
  return __builtin_amdgcn_wmma_f32_16x16x32_f16(false, a, false, b, (short)0, c, false, false);
}

// ---------------------------------------------------------------------------
// KNN: one block per (batch, 16-row tile). 8 waves compute a 16x128 distance
// strip via WMMA Gram tiles into LDS; 16 threads maintain per-row top-15.
// Next chunk's B fragments are prefetched before the barrier so the top-K
// scan hides the global-load latency. Distance matrix never touches HBM.
// CPAD is a template constant (32 or 64) -> branch-free hot loop.
// ---------------------------------------------------------------------------
template<int CPAD>
__global__ void __launch_bounds__(256) knn_kernel(const _Float16* __restrict__ feath,
                                                  const float* __restrict__ sqn,
                                                  int* __restrict__ idxOut) {
  int b = blockIdx.y;
  int rowBase = blockIdx.x * 16;
  const _Float16* F = feath + (size_t)b * NN * CPAD;
  const float* SQ = sqn + b * NN;
  int tid = threadIdx.x;
  int lane = tid & 31;
  int wave = tid >> 5;
  int nloc = lane & 15;
  int hsel = lane >> 4;

  __shared__ float Sd[16][132];
  __shared__ float sqrow[16];
  __shared__ float bd[16][16];
  __shared__ int   bi[16][16];

  if (tid < 16) {
    sqrow[tid] = SQ[rowBase + tid];
    for (int k = 0; k < 16; ++k) { bd[tid][k] = 3.0e38f; bi[tid][k] = 0; }
  }
  __syncthreads();

  // A fragments for this block's 16 rows (reused for all 32 column chunks)
  const _Float16* pa = F + (size_t)(rowBase + nloc) * CPAD + 16*hsel;
  v16h a0 = lda(pa);
  v16h a1 = (CPAD > 32) ? lda(pa + 32) : a0;

  // per-lane B pointer, advanced by 128 rows per chunk
  const _Float16* pb = F + (size_t)(wave*16 + nloc) * CPAD + 16*hsel;
  v16h b0 = ldb(pb);
  v16h b1 = (CPAD > 32) ? ldb(pb + 32) : b0;
  v16h nb0 = b0, nb1 = b1;

  int m0 = hsel * 8;

  for (int cc = 0; cc < NN/128; ++cc) {
    int colBase = cc*128 + wave*16;
    float sc = SQ[colBase + nloc];
    v8f acc = {0.f,0.f,0.f,0.f,0.f,0.f,0.f,0.f};
    acc = wmma(a0, b0, acc);
    if (CPAD > 32) acc = wmma(a1, b1, acc);
    // prefetch next chunk's B fragments (latency hidden by epilogue + scan)
    if (cc < NN/128 - 1) {
      pb += (size_t)128 * CPAD;
      nb0 = ldb(pb);
      if (CPAD > 32) nb1 = ldb(pb + 32);
    }
    #pragma unroll
    for (int v = 0; v < 8; ++v) {
      int m = m0 + v;
      Sd[m][wave*16 + nloc] = sqrow[m] + sc - 2.0f * acc[v];
    }
    __syncthreads();
    if (tid < 16) {
      int t = tid;
      for (int j = 0; j < 128; ++j) {
        float d = Sd[t][j];
        if (d < bd[t][KNBR-1]) {
          int pos = KNBR-1;
          while (pos > 0 && bd[t][pos-1] > d) {
            bd[t][pos] = bd[t][pos-1]; bi[t][pos] = bi[t][pos-1]; --pos;
          }
          bd[t][pos] = d; bi[t][pos] = cc*128 + j;
        }
      }
    }
    __syncthreads();
    b0 = nb0; b1 = nb1;
  }
  if (tid < 16) {
    int gp = b*NN + rowBase + tid;
    for (int k = 0; k < KNBR; ++k) idxOut[gp*KNBR + k] = bi[tid][k];
  }
}

// ---------------------------------------------------------------------------
// EdgeConv layer-1: one wave per 16-edge row tile x all 64 outputs.
// A fragment = [ f[p] | f[q]-f[p] ] built on the fly (fused gather),
// shared by 4 WMMAs; K-loop fully unrolled (CPAD compile-time) so the
// ctr/diff side selection is static and all loads hoist ahead of the WMMAs.
// ---------------------------------------------------------------------------
template<int CPAD>
static __device__ __forceinline__ v8h edge_chunk(const _Float16* fp, const _Float16* fq, int s) {
  if (s < CPAD) return *(const v8h*)(fp + s);
  v8h a = *(const v8h*)(fq + (s - CPAD));
  v8h b = *(const v8h*)(fp + (s - CPAD));
  return a - b;
}
template<int CPAD>
static __device__ __forceinline__ v16h edge_frag(const _Float16* fp, const _Float16* fq,
                                                 int kb, int hsel) {
  v8h lo = edge_chunk<CPAD>(fp, fq, kb + 16*hsel);
  v8h hi = edge_chunk<CPAD>(fp, fq, kb + 16*hsel + 8);
  return cat16(lo, hi);
}

template<int CPAD>
__global__ void __launch_bounds__(256) edge_gemm1(const _Float16* __restrict__ feath,
                                                  const int* __restrict__ knn,
                                                  const _Float16* __restrict__ Wcm,
                                                  const float* __restrict__ bias,
                                                  float* __restrict__ out) {
  const int Nout = 64;
  const int Kpad = 2*CPAD;
  int mt = blockIdx.x * (blockDim.x >> 5) + (threadIdx.x >> 5);
  if (mt >= MEDGE/16) return;
  int lane = threadIdx.x & 31;
  int r = lane & 15, hsel = lane >> 4;
  int e  = mt*16 + r;
  int p  = e / KNBR;
  int q  = (p / NN) * NN + knn[e];
  const _Float16* fp = feath + (size_t)p * CPAD;
  const _Float16* fq = feath + (size_t)q * CPAD;

  const _Float16* pb0 = Wcm + (size_t)(r +  0) * Kpad + 16*hsel;
  const _Float16* pb1 = Wcm + (size_t)(r + 16) * Kpad + 16*hsel;
  const _Float16* pb2 = Wcm + (size_t)(r + 32) * Kpad + 16*hsel;
  const _Float16* pb3 = Wcm + (size_t)(r + 48) * Kpad + 16*hsel;

  v8f ac0 = {0.f,0.f,0.f,0.f,0.f,0.f,0.f,0.f};
  v8f ac1 = ac0, ac2 = ac0, ac3 = ac0;

  #pragma unroll
  for (int kb = 0; kb < Kpad; kb += 32) {
    v16h a  = edge_frag<CPAD>(fp, fq, kb, hsel);
    v16h b0 = ldb(pb0 + kb), b1 = ldb(pb1 + kb), b2 = ldb(pb2 + kb), b3 = ldb(pb3 + kb);
    ac0 = wmma(a, b0, ac0); ac1 = wmma(a, b1, ac1);
    ac2 = wmma(a, b2, ac2); ac3 = wmma(a, b3, ac3);
  }

  int mBase = mt*16 + hsel*8;
  v8f accs[4] = {ac0, ac1, ac2, ac3};
  #pragma unroll
  for (int j = 0; j < 4; ++j) {
    int col = j*16 + r;
    float bv = bias[col];
    #pragma unroll
    for (int v = 0; v < 8; ++v)
      out[(size_t)(mBase + v) * Nout + col] = fmaxf(accs[j][v] + bv, 0.0f);
  }
}

// ---------------------------------------------------------------------------
// Generic WMMA GEMM: out(M,Nout) = relu(A(M,Kpad) @ W + bias)
// One wave per 16x64 tile: 4 accumulators share each A fragment (4 WMMA per
// A load); K-loop software-pipelined. Nout must be a multiple of 64.
// ---------------------------------------------------------------------------
__global__ void __launch_bounds__(256) gemm_bias_relu(const _Float16* __restrict__ A,
                                                      const _Float16* __restrict__ Wcm,
                                                      const float* __restrict__ bias,
                                                      float* __restrict__ out,
                                                      int M, int Nout, int Kpad) {
  int NG = Nout >> 6;
  int total = (M >> 4) * NG;
  int wv = blockIdx.x * (blockDim.x >> 5) + (threadIdx.x >> 5);
  if (wv >= total) return;
  int mt = wv / NG, ng = wv % NG;
  int lane = threadIdx.x & 31;
  int r = lane & 15, hsel = lane >> 4;

  const _Float16* pa  = A + (size_t)(mt*16 + r) * Kpad + 16*hsel;
  const _Float16* pb0 = Wcm + (size_t)(ng*64 + r +  0) * Kpad + 16*hsel;
  const _Float16* pb1 = Wcm + (size_t)(ng*64 + r + 16) * Kpad + 16*hsel;
  const _Float16* pb2 = Wcm + (size_t)(ng*64 + r + 32) * Kpad + 16*hsel;
  const _Float16* pb3 = Wcm + (size_t)(ng*64 + r + 48) * Kpad + 16*hsel;

  v8f ac0 = {0.f,0.f,0.f,0.f,0.f,0.f,0.f,0.f};
  v8f ac1 = ac0, ac2 = ac0, ac3 = ac0;

  v16h a  = lda(pa);
  v16h b0 = ldb(pb0), b1 = ldb(pb1), b2 = ldb(pb2), b3 = ldb(pb3);

  for (int kb = 32; kb < Kpad; kb += 32) {
    v16h an = lda(pa + kb);
    v16h c0 = ldb(pb0 + kb), c1 = ldb(pb1 + kb), c2 = ldb(pb2 + kb), c3 = ldb(pb3 + kb);
    ac0 = wmma(a, b0, ac0); ac1 = wmma(a, b1, ac1);
    ac2 = wmma(a, b2, ac2); ac3 = wmma(a, b3, ac3);
    a = an; b0 = c0; b1 = c1; b2 = c2; b3 = c3;
  }
  ac0 = wmma(a, b0, ac0); ac1 = wmma(a, b1, ac1);
  ac2 = wmma(a, b2, ac2); ac3 = wmma(a, b3, ac3);

  int mBase = mt*16 + hsel*8;
  v8f accs[4] = {ac0, ac1, ac2, ac3};
  #pragma unroll
  for (int j = 0; j < 4; ++j) {
    int col = ng*64 + j*16 + r;
    float bv = bias[col];
    #pragma unroll
    for (int v = 0; v < 8; ++v)
      out[(size_t)(mBase + v) * Nout + col] = fmaxf(accs[j][v] + bv, 0.0f);
  }
}

// ---------------------------------------------------------------------------
// BatchNorm (training mode, stats over axis 0): LDS ds_add_f32 partials ->
// global atomics -> per-channel scale/shift -> apply (+ optional f16 output).
// C is always a power of two (64/256/1024/128).
// ---------------------------------------------------------------------------
__global__ void bn_stats(const float* __restrict__ h, int total, int Cmask,
                         float* __restrict__ sum, float* __restrict__ sumsq) {
  __shared__ float s1[1024];
  __shared__ float s2[1024];
  int C = Cmask + 1;
  for (int c = threadIdx.x; c < C; c += blockDim.x) { s1[c] = 0.f; s2[c] = 0.f; }
  __syncthreads();
  for (size_t i = (size_t)blockIdx.x*blockDim.x + threadIdx.x; i < (size_t)total;
       i += (size_t)gridDim.x*blockDim.x) {
    float v = h[i];
    int c = (int)(i & (size_t)Cmask);
    atomicAdd(&s1[c], v);
    atomicAdd(&s2[c], v*v);
  }
  __syncthreads();
  for (int c = threadIdx.x; c < C; c += blockDim.x) {
    atomicAdd(&sum[c], s1[c]);
    atomicAdd(&sumsq[c], s2[c]);
  }
}

__global__ void bn_finalize(const float* sum, const float* sumsq,
                            const float* g, const float* be,
                            float* scale, float* shift, int C, float invM) {
  int c = blockIdx.x*blockDim.x + threadIdx.x;
  if (c >= C) return;
  float m   = sum[c] * invM;
  float var = fmaxf(sumsq[c] * invM - m*m, 0.0f);
  float sc  = g[c] * rsqrtf(var + BN_EPS);
  scale[c] = sc;
  shift[c] = be[c] - m*sc;
}

__global__ void bn_apply(const float* __restrict__ h, const float* __restrict__ scale,
                         const float* __restrict__ shift, float* __restrict__ of32,
                         _Float16* __restrict__ of16, int total, int Cmask) {
  for (size_t i = (size_t)blockIdx.x*blockDim.x + threadIdx.x; i < (size_t)total;
       i += (size_t)gridDim.x*blockDim.x) {
    int c = (int)(i & (size_t)Cmask);
    float v = fmaf(h[i], scale[c], shift[c]);
    if (of32) of32[i] = v;
    if (of16) of16[i] = (_Float16)v;
  }
}

// ---------------------------------------------------------------------------
// Small elementwise kernels
// ---------------------------------------------------------------------------
__global__ void max_over_k(const float* __restrict__ h, _Float16* __restrict__ xh) {
  int i = blockIdx.x*blockDim.x + threadIdx.x;   // = p*64 + c
  if (i >= NPTS*64) return;
  int p = i >> 6, c = i & 63;
  const float* base = h + (size_t)p * KNBR * 64 + c;
  float m = -3.0e38f;
  #pragma unroll
  for (int k = 0; k < KNBR; ++k) m = fmaxf(m, base[k*64]);
  xh[i] = (_Float16)m;
}

__global__ void sqnorm64(const _Float16* __restrict__ xh, float* __restrict__ sq) {
  int p = blockIdx.x*blockDim.x + threadIdx.x;
  if (p >= NPTS) return;
  float s = 0.f;
  #pragma unroll 8
  for (int c = 0; c < 64; ++c) { float v = (float)xh[(size_t)p*64 + c]; s += v*v; }
  sq[p] = s;
}

__global__ void build_x0(const float* __restrict__ x, const float* __restrict__ pos,
                         _Float16* __restrict__ x0h, float* __restrict__ sq) {
  int p = blockIdx.x*blockDim.x + threadIdx.x;
  if (p >= NPTS) return;
  float s = 0.f;
  for (int c = 0; c < 25; ++c) { float v = x[p*25 + c];  x0h[(size_t)p*32 + c]      = (_Float16)v; s += v*v; }
  for (int c = 0; c < 3;  ++c) { float v = pos[p*3 + c]; x0h[(size_t)p*32 + 25 + c] = (_Float16)v; s += v*v; }
  for (int c = 28; c < 32; ++c) x0h[(size_t)p*32 + c] = (_Float16)0.f;
  sq[p] = s;
}

__global__ void concat3(const _Float16* __restrict__ a, const _Float16* __restrict__ b,
                        const _Float16* __restrict__ c, _Float16* __restrict__ o) {
  int i = blockIdx.x*blockDim.x + threadIdx.x;  // NPTS*192
  if (i >= NPTS*192) return;
  int p = i / 192, cc = i % 192;
  _Float16 v = (cc < 64) ? a[(size_t)p*64 + cc]
             : (cc < 128) ? b[(size_t)p*64 + cc - 64]
                          : c[(size_t)p*64 + cc - 128];
  o[i] = v;
}

// weight (din x dout, f32 row-major) -> f16 column-major padded (dout rows x Kpad)
__global__ void pack_w(const float* __restrict__ W, _Float16* __restrict__ Wcm,
                       int din, int Kpad, int dout) {
  int i = blockIdx.x*blockDim.x + threadIdx.x;
  if (i >= dout*Kpad) return;
  int o = i / Kpad, kk = i % Kpad;
  float v = (kk < din) ? W[(size_t)kk*dout + o] : 0.0f;
  Wcm[(size_t)o*Kpad + kk] = (_Float16)v;
}

// edge layer-1 weights ((2*Creal) x dout) -> f16 CM with per-half padding to Cpad
__global__ void pack_w_edge(const float* __restrict__ W, _Float16* __restrict__ Wcm,
                            int Creal, int Cpad, int dout) {
  int Kpad = 2*Cpad;
  int i = blockIdx.x*blockDim.x + threadIdx.x;
  if (i >= dout*Kpad) return;
  int o = i / Kpad, kk = i % Kpad;
  int hsel = kk / Cpad, r = kk % Cpad;
  float v = (r < Creal) ? W[(size_t)(hsel*Creal + r)*dout + o] : 0.0f;
  Wcm[(size_t)o*Kpad + kk] = (_Float16)v;
}

// final 128->10 + log_softmax, one thread per row
__global__ void final_logsoftmax(const _Float16* __restrict__ h, const float* __restrict__ W,
                                 const float* __restrict__ bias, float* __restrict__ out) {
  int p = blockIdx.x*blockDim.x + threadIdx.x;
  if (p >= NPTS) return;
  float s[10];
  #pragma unroll
  for (int o = 0; o < 10; ++o) s[o] = bias[o];
  const _Float16* hp = h + (size_t)p*128;
  for (int c = 0; c < 128; ++c) {
    float v = (float)hp[c];
    const float* wr = W + c*10;
    #pragma unroll
    for (int o = 0; o < 10; ++o) s[o] = fmaf(v, wr[o], s[o]);
  }
  float mx = s[0];
  #pragma unroll
  for (int o = 1; o < 10; ++o) mx = fmaxf(mx, s[o]);
  float sum = 0.f;
  #pragma unroll
  for (int o = 0; o < 10; ++o) sum += expf(s[o] - mx);
  float l = logf(sum);
  #pragma unroll
  for (int o = 0; o < 10; ++o) out[(size_t)p*10 + o] = s[o] - mx - l;
}

// ---------------------------------------------------------------------------
// Host orchestration
// ---------------------------------------------------------------------------
static void run_bn(const float* h, int M, int C, const float* g, const float* be,
                   float* of32, _Float16* of16,
                   float* sum, float* sumsq, float* scale, float* shift,
                   hipStream_t stream) {
  hipMemsetAsync(sum, 0, 2*1024*sizeof(float), stream);   // zeroes sum+sumsq (contiguous)
  bn_stats<<<1024, 256, 0, stream>>>(h, M*C, C-1, sum, sumsq);
  bn_finalize<<<(C+255)/256, 256, 0, stream>>>(sum, sumsq, g, be, scale, shift, C, 1.0f/(float)M);
  bn_apply<<<4096, 256, 0, stream>>>(h, scale, shift, of32, of16, M*C, C-1);
}

extern "C" void kernel_launch(void* const* d_in, const int* in_sizes, int n_in,
                              void* d_out, int out_size, void* d_ws, size_t ws_size,
                              hipStream_t stream) {
  (void)in_sizes; (void)n_in; (void)out_size;
  auto P = [&](int i) { return (const float*)d_in[i]; };
  // d_in order: x, pos, batch, conv1[0]{W,b,g,be}, conv1[1]{..}, conv2[0..1], conv3[0..1],
  //             lin1[0], mlp1[0], mlp2[0], final{W,b}
  const float* x   = P(0);
  const float* pos = P(1);

  char* ws = (char*)d_ws;
  size_t off = 0;
  auto alloc = [&](size_t bytes) { size_t o = off; off = (off + bytes + 255) & ~(size_t)255; return o; };

  size_t o_stats = alloc(4*1024*sizeof(float));
  size_t o_x0h   = alloc((size_t)NPTS*32*2);
  size_t o_sq    = alloc((size_t)NPTS*4);
  size_t o_idx   = alloc((size_t)MEDGE*4);
  size_t o_x1h   = alloc((size_t)NPTS*64*2);
  size_t o_x2h   = alloc((size_t)NPTS*64*2);
  size_t o_x3h   = alloc((size_t)NPTS*64*2);
  size_t o_h192  = alloc((size_t)NPTS*192*2);
  size_t o_h256h = alloc((size_t)NPTS*256*2);
  size_t o_h128h = alloc((size_t)NPTS*128*2);
  size_t o_wc1a = alloc(64*64*2),  o_wc1b = alloc(64*64*2);
  size_t o_wc2a = alloc(64*128*2), o_wc2b = alloc(64*64*2);
  size_t o_wc3a = alloc(64*128*2), o_wc3b = alloc(64*64*2);
  size_t o_wl1  = alloc((size_t)1024*192*2);
  size_t o_wm1  = alloc((size_t)256*1024*2);
  size_t o_wm2  = alloc((size_t)128*256*2);
  size_t o_mid  = alloc((size_t)NPTS*1024*2);     // h1h (31.5MB) / h1024h (33.5MB)
  size_t o_bigb = alloc((size_t)MEDGE*64*4);      // h1/h2 f32 (62.9MB) / h256 f32
  size_t o_biga = alloc((size_t)NPTS*1024*4);     // h2bn f32 / h1024 f32 (67.1MB) / h128 f32
  if (ws_size < off) return;  // insufficient scratch: bail safely

  float*     sum    = (float*)(ws + o_stats);
  float*     sumsq  = sum + 1024;
  float*     scale  = sum + 2048;
  float*     shift  = sum + 3072;
  _Float16*  x0h    = (_Float16*)(ws + o_x0h);
  float*     sq     = (float*)(ws + o_sq);
  int*       idx    = (int*)(ws + o_idx);
  _Float16*  xh[3]  = {(_Float16*)(ws + o_x1h), (_Float16*)(ws + o_x2h), (_Float16*)(ws + o_x3h)};
  _Float16*  h192h  = (_Float16*)(ws + o_h192);
  _Float16*  h256h  = (_Float16*)(ws + o_h256h);
  _Float16*  h128h  = (_Float16*)(ws + o_h128h);
  _Float16*  midh   = (_Float16*)(ws + o_mid);
  float*     bigb   = (float*)(ws + o_bigb);
  float*     biga   = (float*)(ws + o_biga);

  // --- weight repacks (f32 row-major -> f16 column-major, K padded) ---
  auto gpk = [](int n) { return (n + 255) / 256; };
  pack_w_edge<<<gpk(64*64),  256, 0, stream>>>(P(3),  (_Float16*)(ws+o_wc1a), 28, 32, 64);
  pack_w     <<<gpk(64*64),  256, 0, stream>>>(P(7),  (_Float16*)(ws+o_wc1b), 64, 64, 64);
  pack_w_edge<<<gpk(64*128), 256, 0, stream>>>(P(11), (_Float16*)(ws+o_wc2a), 64, 64, 64);
  pack_w     <<<gpk(64*64),  256, 0, stream>>>(P(15), (_Float16*)(ws+o_wc2b), 64, 64, 64);
  pack_w_edge<<<gpk(64*128), 256, 0, stream>>>(P(19), (_Float16*)(ws+o_wc3a), 64, 64, 64);
  pack_w     <<<gpk(64*64),  256, 0, stream>>>(P(23), (_Float16*)(ws+o_wc3b), 64, 64, 64);
  pack_w<<<gpk(1024*192), 256, 0, stream>>>(P(27), (_Float16*)(ws+o_wl1), 192, 192, 1024);
  pack_w<<<gpk(256*1024), 256, 0, stream>>>(P(31), (_Float16*)(ws+o_wm1), 1024, 1024, 256);
  pack_w<<<gpk(128*256),  256, 0, stream>>>(P(35), (_Float16*)(ws+o_wm2), 256, 256, 128);

  // --- x0 = concat([x, pos]) -> f16 (pad 32) + sqnorm ---
  build_x0<<<gpk(NPTS), 256, 0, stream>>>(x, pos, x0h, sq);

  // --- 3 EdgeConv stages ---
  const _Float16* feat[3]   = {x0h, xh[0], xh[1]};
  const size_t    w1o[3]    = {o_wc1a, o_wc2a, o_wc3a};
  const size_t    w2o[3]    = {o_wc1b, o_wc2b, o_wc3b};
  const int       pbase[3]  = {3, 11, 19};

  int edgeBlocks = ((MEDGE/16) + 7) / 8;   // one wave per 16x64 tile
  for (int s = 0; s < 3; ++s) {
    int pb = pbase[s];
    // KNN (WMMA Gram strips + in-LDS top-15)
    if (s == 0) knn_kernel<32><<<dim3(NN/16, BB), 256, 0, stream>>>(feat[s], sq, idx);
    else        knn_kernel<64><<<dim3(NN/16, BB), 256, 0, stream>>>(feat[s], sq, idx);
    // layer 1: fused gather GEMM -> relu -> h1 f32
    if (s == 0) edge_gemm1<32><<<edgeBlocks, 256, 0, stream>>>(feat[s], idx,
                    (_Float16*)(ws+w1o[s]), P(pb+1), bigb);
    else        edge_gemm1<64><<<edgeBlocks, 256, 0, stream>>>(feat[s], idx,
                    (_Float16*)(ws+w1o[s]), P(pb+1), bigb);
    // BN -> f16 h1h
    run_bn(bigb, MEDGE, 64, P(pb+2), P(pb+3), nullptr, midh, sum, sumsq, scale, shift, stream);
    // layer 2: GEMM -> relu -> h2 f32 (reuse bigb)
    gemm_bias_relu<<<edgeBlocks, 256, 0, stream>>>(midh, (_Float16*)(ws+w2o[s]),
                                                   P(pb+5), bigb, MEDGE, 64, 64);
    // BN -> f32 h2bn (biga)
    run_bn(bigb, MEDGE, 64, P(pb+6), P(pb+7), biga, nullptr, sum, sumsq, scale, shift, stream);
    // max over K -> x_{s+1} f16
    max_over_k<<<gpk(NPTS*64), 256, 0, stream>>>(biga, xh[s]);
    // sqnorm for next stage's KNN
    if (s < 2) sqnorm64<<<gpk(NPTS), 256, 0, stream>>>(xh[s], sq);
  }

  // --- head ---
  concat3<<<gpk(NPTS*192), 256, 0, stream>>>(xh[0], xh[1], xh[2], h192h);

  int tl1 = (NPTS/16) * (1024/64);
  gemm_bias_relu<<<(tl1+7)/8, 256, 0, stream>>>(h192h, (_Float16*)(ws+o_wl1), P(28),
                                                biga, NPTS, 1024, 192);
  run_bn(biga, NPTS, 1024, P(29), P(30), nullptr, midh, sum, sumsq, scale, shift, stream);

  int tm1 = (NPTS/16) * (256/64);
  gemm_bias_relu<<<(tm1+7)/8, 256, 0, stream>>>(midh, (_Float16*)(ws+o_wm1), P(32),
                                                bigb, NPTS, 256, 1024);
  run_bn(bigb, NPTS, 256, P(33), P(34), nullptr, h256h, sum, sumsq, scale, shift, stream);

  int tm2 = (NPTS/16) * (128/64);
  gemm_bias_relu<<<(tm2+7)/8, 256, 0, stream>>>(h256h, (_Float16*)(ws+o_wm2), P(36),
                                                biga, NPTS, 128, 256);
  run_bn(biga, NPTS, 128, P(37), P(38), nullptr, h128h, sum, sumsq, scale, shift, stream);

  final_logsoftmax<<<gpk(NPTS), 256, 0, stream>>>(h128h, P(39), P(40), (float*)d_out);
}
